// HLMGPT2Block_7541962571919
// MI455X (gfx1250) — matrix operand order repaired and checked by reading, not compiled, not run on hardware
//
#include <hip/hip_runtime.h>
#include <math.h>

#define BATCH    2
#define SEQ      2176
#define HID      768
#define HEADS    12
#define HEAD_DIM 64
#define INNER    3072
#define WIN      17
#define NBLK     (SEQ / WIN)          /* 128 */
#define ROWS     (BATCH * SEQ)        /* 4352 */
#define QKV_W    (3 * HID)            /* 2304 */
#define MAXK     160                  /* >= WIN + NBLK = 145 */

typedef __attribute__((ext_vector_type(16))) __bf16 v16bf;
typedef __attribute__((ext_vector_type(8)))  __bf16 v8bf;
typedef __attribute__((ext_vector_type(8)))  float  v8f;

// ---------------------------------------------------------------------------
// Weight convert: W[K][N] fp32 -> Wt[N][K] bf16 (transpose so each WMMA B-lane
// reads 16 K-contiguous bf16 = two b128 loads)
// ---------------------------------------------------------------------------
__global__ __launch_bounds__(256)
void convt_kernel(const float* __restrict__ W, __bf16* __restrict__ Wt,
                  int K, int N) {
    size_t idx = (size_t)blockIdx.x * 256 + threadIdx.x;
    size_t tot = (size_t)K * N;
    if (idx < tot) {
        int k = (int)(idx / N);
        int n = (int)(idx % N);
        Wt[(size_t)n * K + k] = (__bf16)W[idx];
    }
}

// ---------------------------------------------------------------------------
// LayerNorm over HID=768 per row, fp32 in -> bf16 out (GEMM A operand)
// ---------------------------------------------------------------------------
__global__ __launch_bounds__(256)
void ln_kernel(const float* __restrict__ x, const float* __restrict__ g,
               const float* __restrict__ b, __bf16* __restrict__ y) {
    __shared__ float s1[256];
    __shared__ float s2[256];
    int row = blockIdx.x;
    const float* xr = x + (size_t)row * HID;
    float sum = 0.f, sq = 0.f;
    for (int j = threadIdx.x; j < HID; j += 256) {
        float v = xr[j];
        sum += v; sq += v * v;
    }
    s1[threadIdx.x] = sum; s2[threadIdx.x] = sq;
    __syncthreads();
    for (int off = 128; off > 0; off >>= 1) {
        if (threadIdx.x < off) {
            s1[threadIdx.x] += s1[threadIdx.x + off];
            s2[threadIdx.x] += s2[threadIdx.x + off];
        }
        __syncthreads();
    }
    float mean = s1[0] * (1.f / HID);
    float var  = s2[0] * (1.f / HID) - mean * mean;
    float inv  = rsqrtf(var + 1e-5f);
    __bf16* yr = y + (size_t)row * HID;
    for (int j = threadIdx.x; j < HID; j += 256)
        yr[j] = (__bf16)((xr[j] - mean) * inv * g[j] + b[j]);
}

// ---------------------------------------------------------------------------
// WMMA bf16 GEMM: out[M,N] = A[M,K] @ Bt[N,K]^T + bias (+residual) (+gelu)
// One wave computes a 16x64 strip (4 N-tiles) so the A fragment is loaded
// once and reused by 4 WMMAs per K-step (2.5 b128 loads per WMMA).
// 8 waves / block. K%32==0, M%16==0, N%64==0. EXEC all-ones throughout.
// ---------------------------------------------------------------------------
__device__ __forceinline__ float gelu_new_f(float x) {
    const float c = 0.7978845608028654f;          // sqrt(2/pi)
    float x3 = x * x * x;
    return 0.5f * x * (1.f + tanhf(c * (x + 0.044715f * x3)));
}

__global__ __launch_bounds__(256)
void gemm_wmma_kernel(const __bf16* __restrict__ A, const __bf16* __restrict__ Bt,
                      const float* __restrict__ bias,
                      const float* __restrict__ residual,
                      float* __restrict__ outF, __bf16* __restrict__ outB,
                      int M, int N, int K, int gelu_flag) {
    const int lane = threadIdx.x & 31;
    const int wid  = threadIdx.x >> 5;
    const int ntn4 = N >> 6;                 // 64-wide N strips
    const int tile = blockIdx.x * 8 + wid;
    const int tm   = tile / ntn4;
    const int tn4  = tile % ntn4;
    const int half = lane >> 4;              // K-half selector (this code's contiguous k = 16h + s order)
    const int l16  = lane & 15;

    // A: lane l16 owns row m=l16; K elements k0+16h .. k0+16h+15 contiguous
    const __bf16* arow = A + (size_t)(tm * 16 + l16) * K;
    // B: lane l16 owns col n; K elements k0+16h .. k0+16h+15 contiguous
    const __bf16* brow0 = Bt + (size_t)(tn4 * 64 + 0  + l16) * K;
    const __bf16* brow1 = Bt + (size_t)(tn4 * 64 + 16 + l16) * K;
    const __bf16* brow2 = Bt + (size_t)(tn4 * 64 + 32 + l16) * K;
    const __bf16* brow3 = Bt + (size_t)(tn4 * 64 + 48 + l16) * K;

    v8f c0 = {}, c1 = {}, c2 = {}, c3 = {};
    for (int k0 = 0; k0 < K; k0 += 32) {
        v8bf alo = *(const v8bf*)(arow + k0 + 16 * half);
        v8bf ahi = *(const v8bf*)(arow + k0 + 16 * half + 8);
        v16bf af;
#pragma unroll
        for (int i = 0; i < 8; ++i) { af[i] = alo[i]; af[i + 8] = ahi[i]; }

        const int kb = k0 + 16 * half;
        v8bf b0l = *(const v8bf*)(brow0 + kb), b0h = *(const v8bf*)(brow0 + kb + 8);
        v8bf b1l = *(const v8bf*)(brow1 + kb), b1h = *(const v8bf*)(brow1 + kb + 8);
        v8bf b2l = *(const v8bf*)(brow2 + kb), b2h = *(const v8bf*)(brow2 + kb + 8);
        v8bf b3l = *(const v8bf*)(brow3 + kb), b3h = *(const v8bf*)(brow3 + kb + 8);
        v16bf bf0, bf1, bf2, bf3;
#pragma unroll
        for (int i = 0; i < 8; ++i) {
            bf0[i] = b0l[i]; bf0[i + 8] = b0h[i];
            bf1[i] = b1l[i]; bf1[i + 8] = b1h[i];
            bf2[i] = b2l[i]; bf2[i + 8] = b2h[i];
            bf3[i] = b3l[i]; bf3[i + 8] = b3h[i];
        }
        c0 = __builtin_amdgcn_wmma_f32_16x16x32_bf16(false, af, false, bf0, (short)0, c0, false, false);
        c1 = __builtin_amdgcn_wmma_f32_16x16x32_bf16(false, af, false, bf1, (short)0, c1, false, false);
        c2 = __builtin_amdgcn_wmma_f32_16x16x32_bf16(false, af, false, bf2, (short)0, c2, false, false);
        c3 = __builtin_amdgcn_wmma_f32_16x16x32_bf16(false, af, false, bf3, (short)0, c3, false, false);
    }

    v8f acc[4] = {c0, c1, c2, c3};
#pragma unroll
    for (int sub = 0; sub < 4; ++sub) {
        const int n  = tn4 * 64 + sub * 16 + l16;
        const float bv = bias ? bias[n] : 0.f;
#pragma unroll
        for (int r = 0; r < 8; ++r) {
            int m = tm * 16 + r + 8 * half;   // D layout: VGPR r -> M=r (+8 hi half)
            float v = acc[sub][r] + bv;
            if (residual) v += residual[(size_t)m * N + n];
            if (gelu_flag) v = gelu_new_f(v);
            if (outF) outF[(size_t)m * N + n] = v;
            if (outB) outB[(size_t)m * N + n] = (__bf16)v;
        }
    }
}

// ---------------------------------------------------------------------------
// Sparse ctx-pred attention: one wave per (b, h, q). Allowed keys:
//   same window w0..s (causal) plus window heads j*WIN for j < blk.
// nk <= 145. Softmax in fp32, output context as bf16 [row, h*64+d].
// ---------------------------------------------------------------------------
__global__ __launch_bounds__(256)
void attn_kernel(const float* __restrict__ qkv, __bf16* __restrict__ ctx) {
    __shared__ float smem[8 * MAXK];
    const int lane = threadIdx.x & 31;
    const int wid  = threadIdx.x >> 5;
    const int gw   = blockIdx.x * 8 + wid;      // grid sized exactly
    const int h    = gw % HEADS;
    const int row  = gw / HEADS;
    const int b    = row / SEQ;
    const int s    = row % SEQ;
    const int blk  = s / WIN;
    const int w0   = blk * WIN;
    const int nk   = blk + (s - w0) + 1;

    const float* qr = qkv + (size_t)row * QKV_W + h * HEAD_DIM;
    float* sc = smem + wid * MAXK;

    float mx = -3.4e38f;
    for (int t = lane; t < nk; t += 32) {
        int kp = (t < blk) ? t * WIN : w0 + (t - blk);
        const float* kr = qkv + (size_t)(b * SEQ + kp) * QKV_W + HID + h * HEAD_DIM;
        float acc = 0.f;
#pragma unroll 8
        for (int d = 0; d < HEAD_DIM; ++d) acc += qr[d] * kr[d];
        acc *= 0.125f;                          // 1/sqrt(64)
        sc[t] = acc;
        mx = fmaxf(mx, acc);
    }
    for (int off = 16; off > 0; off >>= 1)
        mx = fmaxf(mx, __shfl_xor(mx, off, 32));

    float sum = 0.f;
    for (int t = lane; t < nk; t += 32) {
        float e = __expf(sc[t] - mx);
        sc[t] = e;
        sum += e;
    }
    for (int off = 16; off > 0; off >>= 1)
        sum += __shfl_xor(sum, off, 32);
    const float inv = 1.f / sum;

    // AV: lane owns dims {lane, lane+32}; same-wave DS ordering makes sc[] safe
    float a0 = 0.f, a1 = 0.f;
    for (int t = 0; t < nk; ++t) {
        int kp = (t < blk) ? t * WIN : w0 + (t - blk);
        const float* vr = qkv + (size_t)(b * SEQ + kp) * QKV_W + 2 * HID + h * HEAD_DIM;
        float p = sc[t] * inv;
        a0 += p * vr[lane];
        a1 += p * vr[lane + 32];
    }
    __bf16* o = ctx + (size_t)row * HID + h * HEAD_DIM;
    o[lane]      = (__bf16)a0;
    o[lane + 32] = (__bf16)a1;
}

// ---------------------------------------------------------------------------
extern "C" void kernel_launch(void* const* d_in, const int* in_sizes, int n_in,
                              void* d_out, int out_size, void* d_ws, size_t ws_size,
                              hipStream_t stream) {
    const float* hs          = (const float*)d_in[0];
    const float* ln1_g       = (const float*)d_in[1];
    const float* ln1_b       = (const float*)d_in[2];
    const float* c_attn_w    = (const float*)d_in[3];
    const float* c_attn_b    = (const float*)d_in[4];
    const float* attn_proj_w = (const float*)d_in[5];
    const float* attn_proj_b = (const float*)d_in[6];
    const float* ln2_g       = (const float*)d_in[7];
    const float* ln2_b       = (const float*)d_in[8];
    const float* fc_w        = (const float*)d_in[9];
    const float* fc_b        = (const float*)d_in[10];
    const float* mlp_proj_w  = (const float*)d_in[11];
    const float* mlp_proj_b  = (const float*)d_in[12];
    float* out = (float*)d_out;

    char* base = (char*)d_ws;
    size_t off = 0;
    auto alloc = [&](size_t bytes) -> char* {
        char* p = base + off;
        off = (off + bytes + 255) & ~(size_t)255;
        return p;
    };
    __bf16* wt_qkv   = (__bf16*)alloc((size_t)QKV_W * HID * 2);   // [2304,768]
    __bf16* wt_aproj = (__bf16*)alloc((size_t)HID * HID * 2);     // [768,768]
    __bf16* wt_fc    = (__bf16*)alloc((size_t)INNER * HID * 2);   // [3072,768]
    __bf16* wt_mproj = (__bf16*)alloc((size_t)HID * INNER * 2);   // [768,3072]
    __bf16* act      = (__bf16*)alloc((size_t)ROWS * HID * 2);    // LN out (reused)
    __bf16* ctx      = (__bf16*)alloc((size_t)ROWS * HID * 2);    // attn context
    __bf16* gelu_act = (__bf16*)alloc((size_t)ROWS * INNER * 2);  // gelu(fc) bf16
    float*  qkv      = (float*)alloc((size_t)ROWS * QKV_W * 4);   // [4352,2304]
    float*  hidden   = (float*)alloc((size_t)ROWS * HID * 4);     // post-attn resid
    (void)ws_size; (void)in_sizes; (void)n_in; (void)out_size;

    // 1) weight transpose-converts fp32 -> bf16 [N,K]
    convt_kernel<<<(HID * QKV_W + 255) / 256, 256, 0, stream>>>(c_attn_w,    wt_qkv,   HID,   QKV_W);
    convt_kernel<<<(HID * HID   + 255) / 256, 256, 0, stream>>>(attn_proj_w, wt_aproj, HID,   HID);
    convt_kernel<<<(HID * INNER + 255) / 256, 256, 0, stream>>>(fc_w,        wt_fc,    HID,   INNER);
    convt_kernel<<<(INNER * HID + 255) / 256, 256, 0, stream>>>(mlp_proj_w,  wt_mproj, INNER, HID);

    // 2) LN1
    ln_kernel<<<ROWS, 256, 0, stream>>>(hs, ln1_g, ln1_b, act);

    // 3) QKV GEMM: [4352,768] x [768,2304] (strips: 272 * 36 / 8 = 1224 blocks)
    gemm_wmma_kernel<<<(ROWS / 16) * (QKV_W / 64) / 8, 256, 0, stream>>>(
        act, wt_qkv, c_attn_b, nullptr, qkv, nullptr, ROWS, QKV_W, HID, 0);

    // 4) sparse windowed attention (4352*12 waves / 8 = 6528 blocks)
    attn_kernel<<<ROWS * HEADS / 8, 256, 0, stream>>>(qkv, ctx);

    // 5) attn proj + residual(hidden_states) -> hidden fp32 (272*12/8 = 408)
    gemm_wmma_kernel<<<(ROWS / 16) * (HID / 64) / 8, 256, 0, stream>>>(
        ctx, wt_aproj, attn_proj_b, hs, hidden, nullptr, ROWS, HID, HID, 0);

    // 6) LN2
    ln_kernel<<<ROWS, 256, 0, stream>>>(hidden, ln2_g, ln2_b, act);

    // 7) FC GEMM + fused GELU -> bf16 (272*48/8 = 1632 blocks)
    gemm_wmma_kernel<<<(ROWS / 16) * (INNER / 64) / 8, 256, 0, stream>>>(
        act, wt_fc, fc_b, nullptr, nullptr, gelu_act, ROWS, INNER, HID, 1);

    // 8) MLP proj + bias + residual(hidden) -> d_out fp32 (272*12/8 = 408)
    gemm_wmma_kernel<<<(ROWS / 16) * (HID / 64) / 8, 256, 0, stream>>>(
        gelu_act, wt_mproj, mlp_proj_b, hidden, out, nullptr, ROWS, HID, INNER, 0);
}
